// InhomogeneousTTNImageModel_7739531067659
// MI455X (gfx1250) — matrix operand order repaired and checked by reading, not compiled, not run on hardware
//
#include <hip/hip_runtime.h>
#include <hip/hip_bf16.h>

typedef __attribute__((ext_vector_type(16))) __bf16 v16bf;
typedef __attribute__((ext_vector_type(8)))  __bf16 v8bf;
typedef __attribute__((ext_vector_type(8)))  float  v8f;

// Native f32<->bf16 conversion (gfx1250 has hardware bf16 cvt; RNE rounding).
__device__ __forceinline__ __bf16 tobf(float f)  { return (__bf16)f; }
__device__ __forceinline__ float  frombf(__bf16 b) { return (float)b; }

#define WMMA_BF16(acc, A, B)                                                   \
    (acc) = __builtin_amdgcn_wmma_f32_16x16x32_bf16(                           \
        false, (A), false, (B), (short)0, (acc), false, false)

#define CAT16 0, 1, 2, 3, 4, 5, 6, 7, 8, 9, 10, 11, 12, 13, 14, 15

// Load a pre-swizzled B fragment: 32 lanes x 16 bf16 contiguous (1 KB).
__device__ __forceinline__ v16bf load_frag_sw(const __bf16* frag, int lane) {
    const v8bf* q = (const v8bf*)(frag + lane * 16);
    return __builtin_shufflevector(q[0], q[1], CAT16);
}

// Load an A fragment for one lane from a contiguous bf16 K-row.
// element e<8 : K = ks*32 + half*8 + e ; e>=8 : K = ks*32 + half*8 + 16 + (e-8)
__device__ __forceinline__ v16bf load_frag_a(const __bf16* row, int ks, int half) {
    const v8bf* lo = (const v8bf*)(row + ks * 32 + half * 8);
    const v8bf* hi = (const v8bf*)(row + ks * 32 + half * 8 + 16);
    return __builtin_shufflevector(*lo, *hi, CAT16);
}

// ---------------------------------------------------------------------------
// Repack a row-major f32 [nmat][K][N] weight into swizzled bf16 B-fragments:
// dst[((mat*(N/16)+nt)*(K/32)+ks)*512 + lane*16 + e],
// with  Kidx = ks*32 + (lane>=16)*8 + (e&7) + (e>=8)*16,  Nidx = nt*16 + lane%16.
// ---------------------------------------------------------------------------
__global__ void repack_b_kernel(const float* __restrict__ src,
                                __bf16* __restrict__ dst,
                                int K, int N, int nmat)
{
    int idx = blockIdx.x * 256 + threadIdx.x;
    int per = K * N;
    if (idx >= nmat * per) return;
    int mat  = idx / per;
    int t    = idx - mat * per;
    int frag = t >> 9;          // nt*Ks + ks
    int r    = t & 511;
    int lane = r >> 4;
    int e    = r & 15;
    int Ks   = K >> 5;
    int ks   = frag % Ks;
    int nt   = frag / Ks;
    int kk   = ks * 32 + (lane >> 4) * 8 + (e & 7) + (e >> 3) * 16;
    int nn   = nt * 16 + (lane & 15);
    dst[idx] = tobf(src[(size_t)mat * per + (size_t)kk * N + nn]);
}

// ---------------------------------------------------------------------------
// Kernel 1: patch embed.  One wave per M-tile (16 rows), all 8 N-tiles.
// A gathered from x as 4x float4 row loads per fragment; B pre-swizzled bf16.
// h0 written as bf16 [131072][128].
// ---------------------------------------------------------------------------
__global__ void patch_embed_kernel(const float* __restrict__ x,
                                   const __bf16* __restrict__ Wp_sw,
                                   const float* __restrict__ bp,
                                   const float* __restrict__ pos,
                                   __bf16* __restrict__ h0)
{
    const int tid  = threadIdx.x;
    const int wave = tid >> 5;
    const int lane = tid & 31;
    const int mt   = blockIdx.x * 4 + wave;     // 0..8191
    const int half = lane >> 4;
    const int mm   = lane & 15;
    const int nn   = lane & 15;

    const int mrow = mt * 16 + mm;
    const int b  = mrow >> 8;
    const int p  = mrow & 255;
    const int gy = p >> 4, gx = p & 15;
    const float* xb = x + (size_t)b * 16384 + gy * 1024 + gx * 8;

    v8f acc[8] = {};
#pragma unroll
    for (int ks = 0; ks < 2; ++ks) {
        v16bf a;
        {
            int py0 = ks * 4 + half;            // rows of the 8x8 patch
            int py1 = py0 + 2;
            const float4* r0 = (const float4*)(xb + py0 * 128);
            const float4* r1 = (const float4*)(xb + py1 * 128);
            float4 p0 = r0[0], p1 = r0[1], p2 = r1[0], p3 = r1[1];
            a[0]=tobf(p0.x);  a[1]=tobf(p0.y);  a[2]=tobf(p0.z);  a[3]=tobf(p0.w);
            a[4]=tobf(p1.x);  a[5]=tobf(p1.y);  a[6]=tobf(p1.z);  a[7]=tobf(p1.w);
            a[8]=tobf(p2.x);  a[9]=tobf(p2.y);  a[10]=tobf(p2.z); a[11]=tobf(p2.w);
            a[12]=tobf(p3.x); a[13]=tobf(p3.y); a[14]=tobf(p3.z); a[15]=tobf(p3.w);
        }
#pragma unroll
        for (int nt = 0; nt < 8; ++nt) {
            v16bf bm = load_frag_sw(Wp_sw + ((size_t)nt * 2 + ks) * 512, lane);
            WMMA_BF16(acc[nt], a, bm);
        }
    }

#pragma unroll
    for (int nt = 0; nt < 8; ++nt) {
#pragma unroll
        for (int v = 0; v < 8; ++v) {
            int r  = mt * 16 + v + half * 8;
            int pp = r & 255;
            int c  = nt * 16 + nn;
            h0[(size_t)r * 128 + c] =
                tobf(acc[nt][v] + bp[c] + pos[(size_t)pp * 128 + c]);
        }
    }
}

// ---------------------------------------------------------------------------
// Kernel 2: one CP tree level (bf16 in/out, swizzled weights).
// ---------------------------------------------------------------------------
__global__ void cp_level_kernel(const __bf16* __restrict__ hin,   // [512,npin,128]
                                const __bf16* __restrict__ f_sw,  // [nodes*4][4*4*512]
                                const __bf16* __restrict__ o_sw,  // [nodes][8*2*512]
                                __bf16* __restrict__ hout,        // [512,nodes,128]
                                int grid_in, int npin)
{
    __shared__ float  u_lds[4][16][64];   // 16 KB
    __shared__ __bf16 p_lds[16][64];      //  2 KB

    const int g2    = grid_in >> 1;
    const int nodes = g2 * g2;
    const int node  = blockIdx.x % nodes;
    const int bt    = blockIdx.x / nodes;

    const int tid  = threadIdx.x;
    const int q    = tid >> 5;            // wave == quad position
    const int lane = tid & 31;
    const int half = lane >> 4;
    const int mm   = lane & 15;
    const int nn   = lane & 15;

    const __bf16* on = o_sw + (size_t)node * 8192;
    __builtin_prefetch(on + lane * 16, 0, 1);   // warm L2 for phase 3

    // ---- phase 1: u_q = H_child (16x128) @ f[node][q] (128x64) ------------
    {
        const int i  = node / g2, j = node % g2;
        const int di = q >> 1, dj = q & 1;
        const int cp = (2 * i + di) * grid_in + (2 * j + dj);
        const __bf16* hrow = hin + ((size_t)(bt * 16 + mm) * npin + cp) * 128;
        const __bf16* fq   = f_sw + (size_t)(node * 4 + q) * 8192;

        v8f acc[4] = {};
#pragma unroll
        for (int ks = 0; ks < 4; ++ks) {
            v16bf a = load_frag_a(hrow, ks, half);
#pragma unroll
            for (int nt = 0; nt < 4; ++nt) {
                v16bf bm = load_frag_sw(fq + ((size_t)nt * 4 + ks) * 512, lane);
                WMMA_BF16(acc[nt], a, bm);
            }
        }
#pragma unroll
        for (int nt = 0; nt < 4; ++nt)
#pragma unroll
            for (int v = 0; v < 8; ++v)
                u_lds[q][v + half * 8][nt * 16 + nn] = acc[nt][v];
    }
    __syncthreads();

    // ---- phase 2: p = prod_q u_q  (f32), store bf16 -----------------------
    for (int idx = tid; idx < 16 * 64; idx += 128) {
        int m = idx >> 6, r = idx & 63;
        p_lds[m][r] = tobf(u_lds[0][m][r] * u_lds[1][m][r] *
                           u_lds[2][m][r] * u_lds[3][m][r]);
    }
    __syncthreads();

    // ---- phase 3: out = p (16x64) @ o[node] (64x128) ----------------------
    v8f acc2[2] = {};
#pragma unroll
    for (int ks = 0; ks < 2; ++ks) {
        v16bf a = load_frag_a(&p_lds[mm][0], ks, half);
#pragma unroll
        for (int t = 0; t < 2; ++t) {
            int nt = q * 2 + t;
            v16bf bm = load_frag_sw(on + ((size_t)nt * 2 + ks) * 512, lane);
            WMMA_BF16(acc2[t], a, bm);
        }
    }
#pragma unroll
    for (int t = 0; t < 2; ++t) {
        int nt = q * 2 + t;
#pragma unroll
        for (int v = 0; v < 8; ++v) {
            int brow = bt * 16 + v + half * 8;
            hout[((size_t)brow * nodes + node) * 128 + nt * 16 + nn] =
                tobf(acc2[t][v]);
        }
    }
}

// ---------------------------------------------------------------------------
// Kernel 3: LayerNorm -> head GEMM (16x128 @ 128x512) -> row L2-normalize.
// ---------------------------------------------------------------------------
__global__ void head_kernel(const __bf16* __restrict__ h4,      // [512,128]
                            const float* __restrict__ gamma,
                            const float* __restrict__ beta,
                            const __bf16* __restrict__ Wh_sw,   // [32*4*512]
                            const float* __restrict__ bh,
                            float* __restrict__ out)            // [512,512]
{
    __shared__ __bf16 xn[16][128];    //  4 KB
    __shared__ float  ob[16][512];    // 32 KB
    __shared__ float  rnorm[16];

    const int tid  = threadIdx.x;
    const int wave = tid >> 5;
    const int lane = tid & 31;
    const int bt   = blockIdx.x;      // 0..31

    if (tid < 16) {
        const __bf16* hr = h4 + (size_t)(bt * 16 + tid) * 128;
        float mu = 0.f;
        for (int c = 0; c < 128; ++c) mu += frombf(hr[c]);
        mu *= (1.f / 128.f);
        float var = 0.f;
        for (int c = 0; c < 128; ++c) { float d = frombf(hr[c]) - mu; var += d * d; }
        var *= (1.f / 128.f);
        float inv = rsqrtf(var + 1e-5f);
        for (int c = 0; c < 128; ++c)
            xn[tid][c] = tobf((frombf(hr[c]) - mu) * inv * gamma[c] + beta[c]);
    }
    __syncthreads();

    const int half = lane >> 4;
    const int mm   = lane & 15;
    const int nn   = lane & 15;
    v8f acc[4] = {};
#pragma unroll
    for (int ks = 0; ks < 4; ++ks) {
        v16bf a = load_frag_a(&xn[mm][0], ks, half);
#pragma unroll
        for (int t = 0; t < 4; ++t) {
            int nt = wave * 4 + t;    // 32 N-tiles over 8 waves
            v16bf bm = load_frag_sw(Wh_sw + ((size_t)nt * 4 + ks) * 512, lane);
            WMMA_BF16(acc[t], a, bm);
        }
    }
#pragma unroll
    for (int t = 0; t < 4; ++t) {
        int nt = wave * 4 + t;
#pragma unroll
        for (int v = 0; v < 8; ++v)
            ob[v + half * 8][nt * 16 + nn] = acc[t][v] + bh[nt * 16 + nn];
    }
    __syncthreads();

    if (tid < 16) {
        float s = 0.f;
        for (int c = 0; c < 512; ++c) s += ob[tid][c] * ob[tid][c];
        rnorm[tid] = fmaxf(sqrtf(s), 1e-12f);
    }
    __syncthreads();

    for (int idx = tid; idx < 16 * 512; idx += 256) {
        int r = idx >> 9, c = idx & 511;
        out[(size_t)(bt * 16 + r) * 512 + c] = ob[r][c] / rnorm[r];
    }
}

// ---------------------------------------------------------------------------
extern "C" void kernel_launch(void* const* d_in, const int* in_sizes, int n_in,
                              void* d_out, int out_size, void* d_ws, size_t ws_size,
                              hipStream_t stream)
{
    (void)in_sizes; (void)n_in; (void)out_size; (void)ws_size;
    const float* x     = (const float*)d_in[0];
    const float* Wp    = (const float*)d_in[1];
    const float* bp    = (const float*)d_in[2];
    const float* pos   = (const float*)d_in[3];
    const float* gamma = (const float*)d_in[4];
    const float* beta  = (const float*)d_in[5];
    const float* Wh    = (const float*)d_in[6];
    const float* bh    = (const float*)d_in[7];
    const float* f[4]  = {(const float*)d_in[8],  (const float*)d_in[10],
                          (const float*)d_in[12], (const float*)d_in[14]};
    const float* o[4]  = {(const float*)d_in[9],  (const float*)d_in[11],
                          (const float*)d_in[13], (const float*)d_in[15]};
    float* out = (float*)d_out;

    // ---- workspace layout (bf16 elements) ----
    __bf16* ws = (__bf16*)d_ws;
    size_t off = 0;
    __bf16* h0 = ws + off;  off += (size_t)512 * 256 * 128;
    __bf16* h1 = ws + off;  off += (size_t)512 *  64 * 128;
    __bf16* h2 = ws + off;  off += (size_t)512 *  16 * 128;
    __bf16* h3 = ws + off;  off += (size_t)512 *   4 * 128;
    __bf16* h4 = ws + off;  off += (size_t)512 *   1 * 128;
    __bf16* wp_sw = ws + off;  off += (size_t)64 * 128;         // 8192
    __bf16* wh_sw = ws + off;  off += (size_t)128 * 512;        // 65536
    __bf16* f_sw[4];
    __bf16* o_sw[4];
    const int nodes_l[4] = {64, 16, 4, 1};
    for (int l = 0; l < 4; ++l) { f_sw[l] = ws + off; off += (size_t)nodes_l[l] * 4 * 128 * 64; }
    for (int l = 0; l < 4; ++l) { o_sw[l] = ws + off; off += (size_t)nodes_l[l] * 64 * 128; }

    // ---- repack weights to swizzled bf16 B-fragments ----
    {
        int tot;
        tot = 64 * 128;
        repack_b_kernel<<<(tot + 255) / 256, 256, 0, stream>>>(Wp, wp_sw, 64, 128, 1);
        tot = 128 * 512;
        repack_b_kernel<<<(tot + 255) / 256, 256, 0, stream>>>(Wh, wh_sw, 128, 512, 1);
        for (int l = 0; l < 4; ++l) {
            tot = nodes_l[l] * 4 * 128 * 64;
            repack_b_kernel<<<(tot + 255) / 256, 256, 0, stream>>>(f[l], f_sw[l], 128, 64, nodes_l[l] * 4);
            tot = nodes_l[l] * 64 * 128;
            repack_b_kernel<<<(tot + 255) / 256, 256, 0, stream>>>(o[l], o_sw[l], 64, 128, nodes_l[l]);
        }
    }

    // ---- pipeline ----
    patch_embed_kernel<<<2048, 128, 0, stream>>>(x, wp_sw, bp, pos, h0);

    const __bf16* hin = h0;
    __bf16* houts[4]  = {h1, h2, h3, h4};
    const int  gin[4]  = {16, 8, 4, 2};
    const int  npin[4] = {256, 64, 16, 4};
    for (int l = 0; l < 4; ++l) {
        int nodes = nodes_l[l];
        cp_level_kernel<<<nodes * 32, 128, 0, stream>>>(hin, f_sw[l], o_sw[l],
                                                        houts[l], gin[l], npin[l]);
        hin = houts[l];
    }

    head_kernel<<<32, 256, 0, stream>>>(h4, gamma, beta, wh_sw, bh, out);
}
